// MultiHeadAttention_34462817583715
// MI455X (gfx1250) — matrix operand order repaired and checked by reading, not compiled, not run on hardware
//
#include <hip/hip_runtime.h>

// Problem constants (from reference): B=2, S=2048, D=2048, H=16, HD=128
#define B_  2
#define S_  2048
#define D_  2048
#define H_  16
#define HD_ 128

typedef __bf16 bf16;
typedef __bf16 v8bf  __attribute__((ext_vector_type(8)));
typedef __bf16 v16bf __attribute__((ext_vector_type(16)));
typedef float  v8f   __attribute__((ext_vector_type(8)));

#define NEG_INF (-__builtin_inff())

// ---- async global->LDS copy (CDNA5 GLOBAL_LOAD_ASYNC_TO_LDS_B128) ----------
// Builtin prototype (from clang diagnostic): (v4i AS1*, v4i AS3*, Imm, Imm)
typedef int v4i __attribute__((vector_size(16)));
typedef v4i __attribute__((address_space(1))) as1_v4i;
typedef v4i __attribute__((address_space(3))) as3_v4i;

__device__ __forceinline__ void async_copy_b128(void* lds, const void* g) {
#if __has_builtin(__builtin_amdgcn_global_load_async_to_lds_b128)
  __builtin_amdgcn_global_load_async_to_lds_b128(
      (as1_v4i*)(unsigned long long)g,
      (as3_v4i*)(unsigned)(unsigned long long)lds, 0, 0);
#else
  unsigned l = (unsigned)(unsigned long long)lds;
  asm volatile("global_load_async_to_lds_b128 %0, %1, off" ::"v"(l), "v"(g)
               : "memory");
#endif
}

#if __has_builtin(__builtin_amdgcn_s_wait_asynccnt)
#define WAIT_ASYNC(n) __builtin_amdgcn_s_wait_asynccnt(n)
#else
#define WAIT_ASYNC(n) asm volatile("s_wait_asynccnt %0" ::"n"(n) : "memory")
#endif

__device__ __forceinline__ v8f wmma_bf16(v16bf a, v16bf b, v8f c) {
  return __builtin_amdgcn_wmma_f32_16x16x32_bf16(false, a, false, b, (short)0, c,
                                                 false, false);
}

union V16U { v16bf v; v8bf h[2]; };

// A-matrix 16x32 bf16 fragment (row-major, leading dim `ld` elements).
__device__ __forceinline__ v16bf load_a16x32(const bf16* base, int ld) {
  int lane = threadIdx.x & 31;
  int m = lane & 15;
  int h8 = (lane >> 4) << 3;
  const bf16* p = base + m * ld + h8;
  V16U u;
  u.h[0] = *reinterpret_cast<const v8bf*>(p);
  u.h[1] = *reinterpret_cast<const v8bf*>(p + 16);
  return u.v;
}

// B-matrix 32x16 bf16 fragment (row-major [k][n], leading dim `ld`).
__device__ __forceinline__ v16bf load_b32x16(const bf16* base, int ld) {
  int lane = threadIdx.x & 31;
  const bf16* p = base + (lane & 15) + ((lane >> 4) << 3) * ld;
  v16bf v;
#pragma unroll
  for (int s = 0; s < 8; ++s) {
    v[s] = p[s * ld];
    v[8 + s] = p[(16 + s) * ld];
  }
  return v;
}

// ---------------------------------------------------------------- convert ---
__global__ void k_f32_to_bf16(const float* __restrict__ in, bf16* __restrict__ out,
                              int n) {
  int i = blockIdx.x * blockDim.x + threadIdx.x;
  int stride = gridDim.x * blockDim.x;
  for (; i < n; i += stride) out[i] = (bf16)in[i];
}

// --------------------------------------------------------------- QKV GEMM ---
// C[4096,6144] = Xb @ Wb + bias; scatter q/v -> [B,H,S,HD], k -> [B,H,HD,S]
// (K stored TRANSPOSED so attention's score B-tiles are contiguous copies).
__global__ __launch_bounds__(256) void k_gemm_qkv(
    const bf16* __restrict__ X, const bf16* __restrict__ W,
    const float* __restrict__ bias, bf16* __restrict__ Qo, bf16* __restrict__ Kto,
    bf16* __restrict__ Vo) {
  __shared__ bf16 As[2][128][32];
  __shared__ bf16 Bs[2][32][128];
  const int tid = threadIdx.x;
  const int lane = tid & 31, wave = tid >> 5;
  const int waveM = wave & 3, waveN = wave >> 2;
  const int tm = blockIdx.x, tn = blockIdx.y;

  const int ar = tid >> 1, ac = (tid & 1) << 4;  // A-tile copy slot (32B)
  const int br = tid >> 3, bc = (tid & 7) << 4;  // B-tile copy slot (32B)
  const bf16* Asrc = X + ((size_t)(tm * 128 + ar)) * D_ + ac;
  const bf16* Bsrc = W + (size_t)br * (3 * D_) + tn * 128 + bc;

#define GEMM_ISSUE(buf, k0)                                                   \
  do {                                                                        \
    async_copy_b128(&As[buf][ar][ac], Asrc + (k0));                           \
    async_copy_b128(&As[buf][ar][ac + 8], Asrc + (k0) + 8);                   \
    async_copy_b128(&Bs[buf][br][bc], Bsrc + (size_t)(k0) * (3 * D_));        \
    async_copy_b128(&Bs[buf][br][bc + 8], Bsrc + (size_t)(k0) * (3 * D_) + 8);\
  } while (0)

  v8f acc[2][4] = {};
  GEMM_ISSUE(0, 0);
  const int NIT = D_ / 32;
  for (int it = 0; it < NIT; ++it) {
    const int cur = it & 1;
    if (it + 1 < NIT) {
      GEMM_ISSUE(cur ^ 1, (it + 1) * 32);
      WAIT_ASYNC(4);  // retire stage `it` (async loads complete in order)
    } else {
      WAIT_ASYNC(0);
    }
    __syncthreads();
    v16bf a0 = load_a16x32(&As[cur][waveM * 32][0], 32);
    v16bf a1 = load_a16x32(&As[cur][waveM * 32 + 16][0], 32);
#pragma unroll
    for (int j = 0; j < 4; ++j) {
      v16bf b = load_b32x16(&Bs[cur][0][waveN * 64 + j * 16], 128);
      acc[0][j] = wmma_bf16(a0, b, acc[0][j]);
      acc[1][j] = wmma_bf16(a1, b, acc[1][j]);
    }
    __syncthreads();
  }
#undef GEMM_ISSUE

  const int nlane = lane & 15, mbase = (lane >> 4) << 3;
#pragma unroll
  for (int i = 0; i < 2; ++i)
#pragma unroll
    for (int j = 0; j < 4; ++j)
#pragma unroll
      for (int r = 0; r < 8; ++r) {
        int m = tm * 128 + waveM * 32 + i * 16 + mbase + r;
        int n = tn * 128 + waveN * 64 + j * 16 + nlane;
        float v = acc[i][j][r] + bias[n];
        int b = m >> 11, s = m & (S_ - 1);
        int h = n / (3 * HD_), rem = n % (3 * HD_);
        int t = rem >> 7, d = rem & (HD_ - 1);
        size_t bh = (size_t)(b * H_ + h);
        if (t == 0)
          Qo[(bh * S_ + s) * HD_ + d] = (bf16)(v * 0.08838834764831845f);
        else if (t == 1)
          Kto[(bh * HD_ + d) * S_ + s] = (bf16)v;  // transposed K
        else
          Vo[(bh * S_ + s) * HD_ + d] = (bf16)v;
      }
}

// -------------------------------------------------------------- attention ---
// Flash-style: 64-query tile per WG, causal sweep of 64-key chunks with
// double-buffered async K/V copies.
__global__ __launch_bounds__(256) void k_attn(
    const bf16* __restrict__ Qb, const bf16* __restrict__ Ktb,
    const bf16* __restrict__ Vb, const int* __restrict__ pmask,
    bf16* __restrict__ Val) {
  __shared__ bf16 Qs[64][128];
  __shared__ bf16 Kts[2][128][64];  // already hd-major (transposed in global)
  __shared__ bf16 Vs[2][64][128];
  __shared__ float Ss[64][64];
  __shared__ bf16 Ps[64][64];
  __shared__ float part[4][64];
  __shared__ float m_row[64], l_row[64], fac_row[64];
  __shared__ int padf[64];

  const int tid = threadIdx.x;
  const int lane = tid & 31, wave = tid >> 5;
  const int waveM = wave & 3, waveN = wave >> 2;
  const int nlane = lane & 15, mbase = (lane >> 4) << 3;
  const int qt = blockIdx.x, h = blockIdx.y, b = blockIdx.z;
  const int q0 = qt * 64;
  const size_t bh = (size_t)b * H_ + h;
  const bf16* Qg = Qb + (bh * S_ + q0) * HD_;
  const bf16* Ktg = Ktb + bh * HD_ * S_;  // [hd][s]
  const bf16* Vg = Vb + bh * S_ * HD_;    // [s][hd]

  // copy slots
  const int qr = tid >> 2, qc = (tid & 3) << 5;   // Q/V: 64 rows x 128 cols
  const int kr = tid >> 1, kc = (tid & 1) << 5;   // Kt: 128 rows x 64 cols

#define ATTN_ISSUE(buf, k0)                                                    \
  do {                                                                         \
    async_copy_b128(&Kts[buf][kr][kc], Ktg + (size_t)kr * S_ + (k0) + kc);     \
    async_copy_b128(&Kts[buf][kr][kc + 8], Ktg + (size_t)kr * S_ + (k0) + kc + 8);\
    async_copy_b128(&Vs[buf][qr][qc], Vg + ((size_t)(k0) + qr) * HD_ + qc);    \
    async_copy_b128(&Vs[buf][qr][qc + 8], Vg + ((size_t)(k0) + qr) * HD_ + qc + 8);\
    async_copy_b128(&Vs[buf][qr][qc + 16], Vg + ((size_t)(k0) + qr) * HD_ + qc + 16);\
    async_copy_b128(&Vs[buf][qr][qc + 24], Vg + ((size_t)(k0) + qr) * HD_ + qc + 24);\
  } while (0)

  // Q tile (once) + first K/V chunk
  async_copy_b128(&Qs[qr][qc], Qg + (size_t)qr * HD_ + qc);
  async_copy_b128(&Qs[qr][qc + 8], Qg + (size_t)qr * HD_ + qc + 8);
  async_copy_b128(&Qs[qr][qc + 16], Qg + (size_t)qr * HD_ + qc + 16);
  async_copy_b128(&Qs[qr][qc + 24], Qg + (size_t)qr * HD_ + qc + 24);
  ATTN_ISSUE(0, 0);
  if (tid < 64) {
    m_row[tid] = NEG_INF;
    l_row[tid] = 0.f;
    padf[tid] = pmask[(size_t)b * S_ + q0 + tid];
  }
  v8f oacc[4] = {};
  WAIT_ASYNC(0);
  __syncthreads();

  const int nchunk = qt + 1;
  for (int ch = 0; ch < nchunk; ++ch) {
    const int cur = ch & 1;
    const int k0 = ch * 64;

    // S = Q @ K^T
    v8f sacc[2] = {};
#pragma unroll
    for (int kk = 0; kk < 4; ++kk) {
      v16bf aq = load_a16x32(&Qs[waveM * 16][kk * 32], 128);
#pragma unroll
      for (int jj = 0; jj < 2; ++jj) {
        v16bf bk = load_b32x16(&Kts[cur][kk * 32][waveN * 32 + jj * 16], 64);
        sacc[jj] = wmma_bf16(aq, bk, sacc[jj]);
      }
    }
#pragma unroll
    for (int jj = 0; jj < 2; ++jj)
#pragma unroll
      for (int r = 0; r < 8; ++r) {
        int m = waveM * 16 + mbase + r;
        int n = waveN * 32 + jj * 16 + nlane;
        float s = sacc[jj][r];
        s = ((k0 + n) > (q0 + m)) ? NEG_INF : (padf[m] ? 0.f : s);
        Ss[m][n] = s;
      }

    // prefetch next chunk while softmax + P@V run
    if (ch + 1 < nchunk) ATTN_ISSUE(cur ^ 1, k0 + 64);
    __syncthreads();

    {  // partial row max
      int row = tid & 63, seg = tid >> 6;
      float mx = NEG_INF;
#pragma unroll
      for (int c = 0; c < 16; ++c) mx = fmaxf(mx, Ss[row][seg * 16 + c]);
      part[seg][row] = mx;
    }
    __syncthreads();
    if (tid < 64) {
      float om = m_row[tid], nm = om;
#pragma unroll
      for (int s = 0; s < 4; ++s) nm = fmaxf(nm, part[s][tid]);
      fac_row[tid] = (om == NEG_INF) ? 0.f : __expf(om - nm);
      m_row[tid] = nm;
    }
    __syncthreads();
    {  // exponentiate, write P (bf16), partial sums
      int row = tid & 63, seg = tid >> 6;
      float mr = m_row[row];
      float sum = 0.f;
#pragma unroll
      for (int c = 0; c < 16; ++c) {
        float sv = Ss[row][seg * 16 + c];
        float p = (sv == NEG_INF) ? 0.f : __expf(sv - mr);
        Ps[row][seg * 16 + c] = (bf16)p;
        sum += p;
      }
      part[seg][row] = sum;
    }
    __syncthreads();
    if (tid < 64)
      l_row[tid] = l_row[tid] * fac_row[tid] + part[0][tid] + part[1][tid] +
                   part[2][tid] + part[3][tid];

    // rescale O, then O += P @ V
#pragma unroll
    for (int j = 0; j < 4; ++j)
#pragma unroll
      for (int r = 0; r < 8; ++r) oacc[j][r] *= fac_row[waveM * 16 + mbase + r];
#pragma unroll
    for (int kk = 0; kk < 2; ++kk) {
      v16bf ap = load_a16x32(&Ps[waveM * 16][kk * 32], 64);
#pragma unroll
      for (int j = 0; j < 4; ++j) {
        v16bf bv = load_b32x16(&Vs[cur][kk * 32][waveN * 64 + j * 16], 128);
        oacc[j] = wmma_bf16(ap, bv, oacc[j]);
      }
    }
    WAIT_ASYNC(0);  // next chunk's copies landed
    __syncthreads();
  }
#undef ATTN_ISSUE

#pragma unroll
  for (int j = 0; j < 4; ++j)
#pragma unroll
    for (int r = 0; r < 8; ++r) {
      int m = waveM * 16 + mbase + r;
      int col = waveN * 64 + j * 16 + nlane;
      float ov = oacc[j][r] / l_row[m];
      Val[((size_t)b * S_ + q0 + m) * D_ + h * HD_ + col] = (bf16)ov;
    }
}

// ------------------------------------------------------------ output GEMM ---
__global__ __launch_bounds__(256) void k_gemm_out(
    const bf16* __restrict__ A, const bf16* __restrict__ W,
    const float* __restrict__ bias, float* __restrict__ out) {
  __shared__ bf16 As[2][128][32];
  __shared__ bf16 Bs[2][32][128];
  const int tid = threadIdx.x;
  const int lane = tid & 31, wave = tid >> 5;
  const int waveM = wave & 3, waveN = wave >> 2;
  const int tm = blockIdx.x, tn = blockIdx.y;

  const int ar = tid >> 1, ac = (tid & 1) << 4;
  const int br = tid >> 3, bc = (tid & 7) << 4;
  const bf16* Asrc = A + ((size_t)(tm * 128 + ar)) * D_ + ac;
  const bf16* Bsrc = W + (size_t)br * D_ + tn * 128 + bc;

#define GEMM_ISSUE(buf, k0)                                                   \
  do {                                                                        \
    async_copy_b128(&As[buf][ar][ac], Asrc + (k0));                           \
    async_copy_b128(&As[buf][ar][ac + 8], Asrc + (k0) + 8);                   \
    async_copy_b128(&Bs[buf][br][bc], Bsrc + (size_t)(k0) * D_);              \
    async_copy_b128(&Bs[buf][br][bc + 8], Bsrc + (size_t)(k0) * D_ + 8);      \
  } while (0)

  v8f acc[2][4] = {};
  GEMM_ISSUE(0, 0);
  const int NIT = D_ / 32;
  for (int it = 0; it < NIT; ++it) {
    const int cur = it & 1;
    if (it + 1 < NIT) {
      GEMM_ISSUE(cur ^ 1, (it + 1) * 32);
      WAIT_ASYNC(4);
    } else {
      WAIT_ASYNC(0);
    }
    __syncthreads();
    v16bf a0 = load_a16x32(&As[cur][waveM * 32][0], 32);
    v16bf a1 = load_a16x32(&As[cur][waveM * 32 + 16][0], 32);
#pragma unroll
    for (int j = 0; j < 4; ++j) {
      v16bf b = load_b32x16(&Bs[cur][0][waveN * 64 + j * 16], 128);
      acc[0][j] = wmma_bf16(a0, b, acc[0][j]);
      acc[1][j] = wmma_bf16(a1, b, acc[1][j]);
    }
    __syncthreads();
  }
#undef GEMM_ISSUE

  const int nlane = lane & 15, mbase = (lane >> 4) << 3;
#pragma unroll
  for (int i = 0; i < 2; ++i)
#pragma unroll
    for (int j = 0; j < 4; ++j)
#pragma unroll
      for (int r = 0; r < 8; ++r) {
        int m = tm * 128 + waveM * 32 + i * 16 + mbase + r;
        int n = tn * 128 + waveN * 64 + j * 16 + nlane;
        out[(size_t)m * D_ + n] = acc[i][j][r] + bias[n];
      }
}

// ------------------------------------------------------------------ launch ---
extern "C" void kernel_launch(void* const* d_in, const int* in_sizes, int n_in,
                              void* d_out, int out_size, void* d_ws,
                              size_t ws_size, hipStream_t stream) {
  (void)in_sizes; (void)n_in; (void)out_size; (void)ws_size;
  const float* x     = (const float*)d_in[0];
  const int*   pmask = (const int*)d_in[1];
  const float* Wqkv  = (const float*)d_in[2];
  const float* bqkv  = (const float*)d_in[3];
  const float* Wout  = (const float*)d_in[4];
  const float* bout  = (const float*)d_in[5];
  float* out = (float*)d_out;

  char* w = (char*)d_ws;
  bf16* xb    = (bf16*)w; w += (size_t)B_ * S_ * D_ * sizeof(bf16);
  bf16* wqkvb = (bf16*)w; w += (size_t)D_ * 3 * D_ * sizeof(bf16);
  bf16* woutb = (bf16*)w; w += (size_t)D_ * D_ * sizeof(bf16);
  bf16* qb    = (bf16*)w; w += (size_t)B_ * H_ * S_ * HD_ * sizeof(bf16);
  bf16* ktb   = (bf16*)w; w += (size_t)B_ * H_ * S_ * HD_ * sizeof(bf16);
  bf16* vb    = (bf16*)w; w += (size_t)B_ * H_ * S_ * HD_ * sizeof(bf16);
  bf16* valb  = (bf16*)w; w += (size_t)B_ * S_ * D_ * sizeof(bf16);

  k_f32_to_bf16<<<2048, 256, 0, stream>>>(x, xb, B_ * S_ * D_);
  k_f32_to_bf16<<<2048, 256, 0, stream>>>(Wqkv, wqkvb, D_ * 3 * D_);
  k_f32_to_bf16<<<2048, 256, 0, stream>>>(Wout, woutb, D_ * D_);

  k_gemm_qkv<<<dim3(32, 48), 256, 0, stream>>>(xb, wqkvb, bqkv, qb, ktb, vb);
  k_attn<<<dim3(32, 16, 2), 256, 0, stream>>>(qb, ktb, vb, pmask, valb);
  k_gemm_out<<<dim3(32, 16), 256, 0, stream>>>(valb, woutb, bout, out);
}
